// Attention_42700564857309
// MI455X (gfx1250) — matrix operand order repaired and checked by reading, not compiled, not run on hardware
//
#include <hip/hip_runtime.h>
#include <hip/hip_bf16.h>
#include <stdint.h>

typedef __attribute__((ext_vector_type(16))) __bf16 v16bf;
typedef __attribute__((ext_vector_type(8)))  float  v8f;

union BfOp { uint32_t u[8]; uint4 q[2]; v16bf v; };

#define NN 4096
#define CDIM 256
#define ADIM 128
#define HD 32
#define NT (NN / 32)
// softmax scale folded with log2(e) so we can use raw v_exp_f32 (exp2)
#define SCALE_L2E (0.17677669529663687f * 1.4426950408889634f)

// ---- bf16 pack: +0x8000 round-to-nearest, high-half pack via v_perm_b32 ----
__device__ __forceinline__ uint32_t packbf(float a, float b) {
    uint32_t ua = __float_as_uint(a) + 0x8000u;
    uint32_t ub = __float_as_uint(b) + 0x8000u;
    // result bytes: [0,1]=ua[2,3], [2,3]=ub[2,3]
    return __builtin_amdgcn_perm(ub, ua, 0x07060302u);
}
__device__ __forceinline__ uint16_t f2bf1(float f) {
    return (uint16_t)((__float_as_uint(f) + 0x8000u) >> 16);
}

#if __has_builtin(__builtin_amdgcn_exp2f)
__device__ __forceinline__ float fexp2(float x) { return __builtin_amdgcn_exp2f(x); }
#else
__device__ __forceinline__ float fexp2(float x) { return __expf(x * 0.6931471805599453f); }
#endif

__device__ __forceinline__ v8f wmma_bf16(const BfOp& a, const BfOp& b, v8f c) {
    return __builtin_amdgcn_wmma_f32_16x16x32_bf16(false, a.v, false, b.v,
                                                   (short)0, c, false, false);
}

// LDS byte-offset of a shared-memory pointer (for async-to-LDS dst)
__device__ __forceinline__ uint32_t lds_off(const void* p) {
    return (uint32_t)(uintptr_t)(__attribute__((address_space(3))) const void*)p;
}
// CDNA5 async global->LDS, per-lane 16B, tracked by ASYNCcnt
__device__ __forceinline__ void async_b128(uint32_t ldsoff, const void* g) {
    unsigned long long ga = (unsigned long long)(uintptr_t)g;
    asm volatile("global_load_async_to_lds_b128 %0, %1, off"
                 :: "v"(ldsoff), "v"(ga) : "memory");
}

// ---------------------------------------------------------------------------
// Kernel 1: transpose + convert weights to bf16.
// ws(u16): [0..32767]=WqT[128][256], [32768..]=WkT, [65536..]=WvT,
//          [98304..131071]=WoT[256][128].
// ---------------------------------------------------------------------------
__global__ void prep_w(const float* __restrict__ Wq, const float* __restrict__ Wk,
                       const float* __restrict__ Wv, const float* __restrict__ Wo,
                       uint16_t* __restrict__ wt) {
    int i = blockIdx.x * blockDim.x + threadIdx.x;   // 131072 threads
    if (i < 3 * 32768) {
        int mat = i >> 15, r = i & 32767;
        int o = r >> 8, in_ = r & 255;
        const float* W = (mat == 0) ? Wq : (mat == 1) ? Wk : Wv;
        wt[i] = f2bf1(W[(size_t)in_ * ADIM + o]);
    } else {
        int r = i - 3 * 32768;
        int o = r >> 7, in_ = r & 127;
        wt[i] = f2bf1(Wo[(size_t)in_ * CDIM + o]);
    }
}

// ---------------------------------------------------------------------------
// Kernel 2: QKV projection. Wave = 16-row strip x 8 col-tiles; K=256.
// Q scaled by SCALE_L2E.  Q,K stored row-major [8192][128]; V stored
// TRANSPOSED as vT[128][8192] for row-contiguous async staging in flash.
// ---------------------------------------------------------------------------
__global__ __launch_bounds__(32) void proj_qkv(const float* __restrict__ query,
                                               const float* __restrict__ context,
                                               const uint16_t* __restrict__ wt,
                                               uint16_t* __restrict__ qkv) {
    const int mat  = blockIdx.y;
    const float* X = (mat == 0) ? query : context;
    const uint16_t* w = wt + (size_t)mat * 128 * 256;
    uint16_t* dst = qkv + (size_t)mat * 8192 * 128;

    const int rt   = blockIdx.x;
    const int lane = threadIdx.x & 31;
    const int n    = lane & 15, hi = lane >> 4;
    const int kbA  = hi ? 8 : 0;
    const int kbB  = hi ? 16 : 0;

    const float* xrow = X + (size_t)(rt * 16 + n) * CDIM;

    v8f acc[8];
#pragma unroll
    for (int t = 0; t < 8; ++t) acc[t] = v8f{};

#pragma unroll
    for (int kk = 0; kk < 8; ++kk) {
        BfOp A;
        {
            const float4* xa = (const float4*)(xrow + 32 * kk + kbA);
            const float4* xb = (const float4*)(xrow + 32 * kk + 16 + kbA);
            float4 a0 = xa[0], a1 = xa[1], b0 = xb[0], b1 = xb[1];
            A.u[0] = packbf(a0.x, a0.y); A.u[1] = packbf(a0.z, a0.w);
            A.u[2] = packbf(a1.x, a1.y); A.u[3] = packbf(a1.z, a1.w);
            A.u[4] = packbf(b0.x, b0.y); A.u[5] = packbf(b0.z, b0.w);
            A.u[6] = packbf(b1.x, b1.y); A.u[7] = packbf(b1.z, b1.w);
        }
#pragma unroll
        for (int ct = 0; ct < 8; ++ct) {
            BfOp B;
            const uint16_t* wr = w + (size_t)(ct * 16 + n) * 256 + 32 * kk + kbA;
            B.q[0] = *(const uint4*)(wr);
            B.q[1] = *(const uint4*)(wr + 16);
            acc[ct] = wmma_bf16(A, B, acc[ct]);
        }
    }

    const float sc = (mat == 0) ? SCALE_L2E : 1.0f;
#pragma unroll
    for (int ct = 0; ct < 8; ++ct)
#pragma unroll
        for (int r = 0; r < 8; ++r) {
            int row = rt * 16 + r + 8 * hi;           // token index
            int col = ct * 16 + n;                    // attn-dim index
            if (mat == 2)
                dst[(size_t)col * 8192 + row] = f2bf1(acc[ct][r]);        // V^T
            else
                dst[(size_t)row * ADIM + col] = f2bf1(acc[ct][r] * sc);   // Q,K
        }
}

// ---------------------------------------------------------------------------
// Kernel 3: fused flash attention, double-buffered async K/V^T staging.
// Block = 4 waves; wave owns a 16-query tile; S^T orientation so softmax
// key-reduction is in-lane + one shfl_xor(16). exp2-domain online softmax.
// ---------------------------------------------------------------------------
__global__ __launch_bounds__(128) void flash_attn(const uint16_t* __restrict__ qw,
                                                  const uint16_t* __restrict__ kw,
                                                  const uint16_t* __restrict__ vT,
                                                  float* __restrict__ ow) {
    __shared__ __align__(16) uint32_t k_lds[2][32][16];   // [buf][key][d-pair]
    __shared__ __align__(16) uint32_t v_lds[2][32][16];   // [buf][d][key-pair]
    __shared__ __align__(16) uint16_t p16[4][16][32];     // per-wave P [q][key]

    const int tid  = threadIdx.x;
    const int lane = tid & 31, wv = tid >> 5;
    const int n    = lane & 15, hi = lane >> 4;
    const int kbA  = hi ? 8 : 0;
    const int kbB  = hi ? 16 : 0;
    const int bh   = blockIdx.y;
    const int b    = bh >> 2, hd = bh & 3;
    const int q0   = blockIdx.x * 64 + wv * 16;

    // per-thread staging addresses (16B chunk each; block covers 2KB/tile)
    const int srow = tid >> 2, sc = tid & 3;
    const uint16_t* kp = kw + ((size_t)(b * NN) + srow) * ADIM + hd * HD + sc * 8;
    const uint16_t* vp = vT + ((size_t)(hd * HD + srow)) * 8192 + b * NN + sc * 8;
    const uint32_t koff = lds_off(&k_lds[0][srow][sc * 4]);
    const uint32_t voff = lds_off(&v_lds[0][srow][sc * 4]);

    // B_Q operand (resident for the whole loop)
    BfOp Bq;
    {
        const uint16_t* qrow = qw + ((size_t)(b * NN + q0 + n)) * ADIM + hd * HD + kbA;
        Bq.q[0] = *(const uint4*)(qrow);
        Bq.q[1] = *(const uint4*)(qrow + 16);
    }

    v8f o0 = v8f{}, o1 = v8f{};
    float mrun = -1e30f, lsum = 0.0f;

    // prefetch tile 0 into buffer 0
    async_b128(koff, kp);
    async_b128(voff, vp);

    for (int jt = 0; jt < NT; ++jt) {
        const int cur = jt & 1;

        __syncthreads();   // everyone done with the buffer we are about to refill
        if (jt + 1 < NT) {
            kp += 32 * ADIM;
            vp += 32;
            const uint32_t boff = (uint32_t)((jt + 1) & 1) * 2048u;
            async_b128(koff + boff, kp);
            async_b128(voff + boff, vp);
            asm volatile("s_wait_asynccnt 0x2" ::: "memory");   // tile jt landed
        } else {
            asm volatile("s_wait_asynccnt 0x0" ::: "memory");
        }
        __syncthreads();   // tile jt visible block-wide

        // ---- S^T = K_tile x Q^T : two WMMAs ----
        BfOp Ak0, Ak1;
        Ak0.q[0] = *(const uint4*)&k_lds[cur][n][kbA >> 1];
        Ak0.q[1] = *(const uint4*)&k_lds[cur][n][8 + (kbA >> 1)];
        Ak1.q[0] = *(const uint4*)&k_lds[cur][16 + n][kbA >> 1];
        Ak1.q[1] = *(const uint4*)&k_lds[cur][16 + n][8 + (kbA >> 1)];
        v8f z = v8f{};
        v8f s0 = wmma_bf16(Ak0, Bq, z);    // (key r+8hi,    q n), log2 domain
        v8f s1 = wmma_bf16(Ak1, Bq, z);    // (key 16+r+8hi, q n)

        // ---- online softmax stats (per query column n) ----
        float tmax = s0[0];
#pragma unroll
        for (int r = 0; r < 8; ++r) {
            tmax = fmaxf(tmax, s0[r]);
            tmax = fmaxf(tmax, s1[r]);
        }
        tmax = fmaxf(tmax, __shfl_xor(tmax, 16, 32));
        const float mnew = fmaxf(mrun, tmax);
        const float corr = fexp2(mrun - mnew);
        float p0[8], p1[8], ts = 0.0f;
#pragma unroll
        for (int r = 0; r < 8; ++r) {
            p0[r] = fexp2(s0[r] - mnew);
            p1[r] = fexp2(s1[r] - mnew);
            ts += p0[r] + p1[r];
        }
#pragma unroll
        for (int r = 0; r < 8; r += 2) {   // packed dword stores into P bounce
            *(uint32_t*)&p16[wv][n][r + 8 * hi]      = packbf(p0[r], p0[r + 1]);
            *(uint32_t*)&p16[wv][n][16 + r + 8 * hi] = packbf(p1[r], p1[r + 1]);
        }
        ts += __shfl_xor(ts, 16, 32);
        lsum = lsum * corr + ts;
        mrun = mnew;

        // ---- rescale accumulators by corr[q], q = r+8hi ----
#pragma unroll
        for (int r = 0; r < 8; ++r) {
            float cw = __shfl(corr, r + 8 * hi, 32);
            o0[r] *= cw;
            o1[r] *= cw;
        }

        // ---- O += P x V ----
        BfOp Ap, Bv0, Bv1;
        Ap.q[0]  = *(const uint4*)&p16[wv][n][kbB];
        Ap.q[1]  = *(const uint4*)&p16[wv][n][8 + kbB];
        Bv0.q[0] = *(const uint4*)&v_lds[cur][n][kbB >> 1];
        Bv0.q[1] = *(const uint4*)&v_lds[cur][n][4 + (kbB >> 1)];
        Bv1.q[0] = *(const uint4*)&v_lds[cur][16 + n][kbB >> 1];
        Bv1.q[1] = *(const uint4*)&v_lds[cur][16 + n][4 + (kbB >> 1)];
        o0 = wmma_bf16(Ap, Bv0, o0);
        o1 = wmma_bf16(Ap, Bv1, o1);
    }

    // ---- normalize and store O (fp32) ----
    const float inv = 1.0f / lsum;
#pragma unroll
    for (int r = 0; r < 8; ++r) {
        float iw = __shfl(inv, r + 8 * hi, 32);
        int row = q0 + r + 8 * hi;
        size_t base = ((size_t)(b * NN + row)) * ADIM + hd * HD;
        ow[base + n]      = o0[r] * iw;
        ow[base + 16 + n] = o1[r] * iw;
    }
}

// ---------------------------------------------------------------------------
// Kernel 4: output projection  out = O[8192x128] @ Wo[128x256]  (fp32 out).
// ---------------------------------------------------------------------------
__global__ __launch_bounds__(32) void proj_out(const float* __restrict__ ow,
                                               const uint16_t* __restrict__ woT,
                                               float* __restrict__ out) {
    const int rt   = blockIdx.x;
    const int half = blockIdx.y;
    const int lane = threadIdx.x & 31;
    const int n    = lane & 15, hi = lane >> 4;
    const int kbA  = hi ? 8 : 0;
    const int kbB  = hi ? 16 : 0;

    const float* xrow = ow + (size_t)(rt * 16 + n) * ADIM;

    v8f acc[8];
#pragma unroll
    for (int t = 0; t < 8; ++t) acc[t] = v8f{};

#pragma unroll
    for (int kk = 0; kk < 4; ++kk) {
        BfOp A;
        {
            const float4* xa = (const float4*)(xrow + 32 * kk + kbA);
            const float4* xb = (const float4*)(xrow + 32 * kk + 16 + kbA);
            float4 a0 = xa[0], a1 = xa[1], b0 = xb[0], b1 = xb[1];
            A.u[0] = packbf(a0.x, a0.y); A.u[1] = packbf(a0.z, a0.w);
            A.u[2] = packbf(a1.x, a1.y); A.u[3] = packbf(a1.z, a1.w);
            A.u[4] = packbf(b0.x, b0.y); A.u[5] = packbf(b0.z, b0.w);
            A.u[6] = packbf(b1.x, b1.y); A.u[7] = packbf(b1.z, b1.w);
        }
#pragma unroll
        for (int ct = 0; ct < 8; ++ct) {
            BfOp B;
            const uint16_t* wr = woT + (size_t)(half * 128 + ct * 16 + n) * 128 + 32 * kk + kbA;
            B.q[0] = *(const uint4*)(wr);
            B.q[1] = *(const uint4*)(wr + 16);
            acc[ct] = wmma_bf16(A, B, acc[ct]);
        }
    }

#pragma unroll
    for (int ct = 0; ct < 8; ++ct)
#pragma unroll
        for (int r = 0; r < 8; ++r) {
            int row = rt * 16 + r + 8 * hi;
            out[(size_t)row * CDIM + half * 128 + ct * 16 + n] = acc[ct][r];
        }
}

// ---------------------------------------------------------------------------
// Workspace map (bytes):
//   [0,        262144)  weights bf16 (WqT,WkT,WvT,WoT)
//   [262144,  2359296)  Q bf16  [8192][128]
//   [2359296, 4456448)  K bf16  [8192][128]
//   [4456448, 6553600)  V^T bf16 [128][8192]
//   [6553600,10747904)  O fp32  [8192][128]
// ---------------------------------------------------------------------------
extern "C" void kernel_launch(void* const* d_in, const int* in_sizes, int n_in,
                              void* d_out, int out_size, void* d_ws, size_t ws_size,
                              hipStream_t stream) {
    const float* query   = (const float*)d_in[0];
    const float* context = (const float*)d_in[1];
    const float* Wq      = (const float*)d_in[2];
    const float* Wk      = (const float*)d_in[3];
    const float* Wv      = (const float*)d_in[4];
    const float* Wo      = (const float*)d_in[5];

    uint8_t*  ws  = (uint8_t*)d_ws;
    uint16_t* wt  = (uint16_t*)ws;
    uint16_t* qkv = (uint16_t*)(ws + 262144);
    float*    ow  = (float*)(ws + 6553600);
    float*    out = (float*)d_out;

    prep_w    <<<dim3(512), 256, 0, stream>>>(Wq, Wk, Wv, Wo, wt);
    proj_qkv  <<<dim3(512, 3), 32, 0, stream>>>(query, context, wt, qkv);
    flash_attn<<<dim3(64, 8), 128, 0, stream>>>(qkv, qkv + 1048576, qkv + 2 * 1048576, ow);
    proj_out  <<<dim3(512, 2), 32, 0, stream>>>(ow, wt + 98304, out);
}
